// MultiHeadAttention_82884278879149
// MI455X (gfx1250) — matrix operand order repaired and checked by reading, not compiled, not run on hardware
//
#include <hip/hip_runtime.h>
#include <stdint.h>

typedef __attribute__((ext_vector_type(16))) __bf16 v16bf;
typedef __attribute__((ext_vector_type(8)))  float  v8f;

#define DEV __device__ __forceinline__

constexpr int B = 8, C = 512, T = 1024, H = 8, DK = 64;
constexpr float NEG_BIG = -3.0e38f;   // stands in for -inf / FILL after masking

DEV unsigned short f2bf(float f) {
  unsigned int u = __float_as_uint(f);
  u += 0x7fffu + ((u >> 16) & 1u);    // round-to-nearest-even
  return (unsigned short)(u >> 16);
}

union Frag16 { v16bf v; uint4 q[2]; unsigned short u[16]; };

// A fragment: 16x32 bf16 (MxK) from tile row-major [M][ld]: one 32B run.
DEV v16bf load_A_frag(const unsigned short* t, int ld) {
  int lane = threadIdx.x & 31;
  int m = lane & 15, hf = lane >> 4;
  Frag16 r;
  r.q[0] = *(const uint4*)(t + m * ld + hf * 16);
  r.q[1] = *(const uint4*)(t + m * ld + hf * 16 + 8);
  return r.v;
}

// B fragment: 32x16 bf16 (KxN) from tile stored TRANSPOSED [N][ld]: one 32B run.
DEV v16bf load_BT_frag(const unsigned short* t, int ld) {
  int lane = threadIdx.x & 31;
  int n = lane & 15, hf = lane >> 4;
  Frag16 r;
  const uint4* p = (const uint4*)(t + n * ld + hf * 16);
  r.q[0] = p[0];
  r.q[1] = p[1];
  return r.v;
}

#define WMMA_BF16(a, b, c) \
  __builtin_amdgcn_wmma_f32_16x16x32_bf16(false, (a), false, (b), (short)0, (c), false, false)

// CDNA5 async copy: global -> LDS, 16 bytes, tracked by ASYNCcnt.
DEV void async_ld_b128(unsigned lds_off, const void* gaddr) {
  asm volatile("global_load_async_to_lds_b128 %0, %1, off"
               :: "v"(lds_off), "v"(gaddr) : "memory");
}
DEV void wait_async0() { asm volatile("s_wait_asynccnt 0x0" ::: "memory"); }
DEV unsigned lds_off32(const void* p) { return (unsigned)(uintptr_t)p; }

// ---------------------------------------------------------------------------
// BN fold: scale = g*rsqrt(var+eps), shift = beta + (bias-mean)*scale
// ---------------------------------------------------------------------------
__global__ void bn_fold(const float* __restrict__ bias, const float* __restrict__ gam,
                        const float* __restrict__ bet,  const float* __restrict__ mean,
                        const float* __restrict__ var,
                        float* __restrict__ scale, float* __restrict__ shift) {
  int i = blockIdx.x * 256 + threadIdx.x;
  if (i >= 4 * C) return;
  float inv = gam[i] * rsqrtf(var[i] + 1e-5f);
  scale[i] = inv;
  shift[i] = bet[i] + (bias[i] - mean[i]) * inv;
}

// ---------------------------------------------------------------------------
// mask bit-pack: int32 (B,1,T,T) -> bits[b][t1][s/32]
// ---------------------------------------------------------------------------
__global__ void mask_pack(const int* __restrict__ mask, unsigned int* __restrict__ bits) {
  int idx = blockIdx.x * 256 + threadIdx.x;
  if (idx >= B * T * (T / 32)) return;
  int wd = idx & 31;
  int t1 = (idx >> 5) & (T - 1);
  int b  = idx >> 15;
  const int4* m4 = (const int4*)(mask + ((size_t)b * T + t1) * T + wd * 32);
  unsigned int bv = 0;
#pragma unroll
  for (int j = 0; j < 8; ++j) {
    int4 v = m4[j];
    bv |= (v.x != 0 ? 1u : 0u) << (4 * j + 0);
    bv |= (v.y != 0 ? 1u : 0u) << (4 * j + 1);
    bv |= (v.z != 0 ? 1u : 0u) << (4 * j + 2);
    bv |= (v.w != 0 ? 1u : 0u) << (4 * j + 3);
  }
  bits[idx] = bv;
}

// ---------------------------------------------------------------------------
// cvt_w: fp32 weights -> bf16, layout-preserving (done once, coalesced)
// ---------------------------------------------------------------------------
__global__ void cvt_w(const float* __restrict__ W, unsigned short* __restrict__ Wb) {
  int idx = blockIdx.x * 256 + threadIdx.x;     // over float4s
  if (idx >= 4 * C * C / 4) return;
  float4 v = *(const float4*)(W + (size_t)idx * 4);
  union { ushort4 s4; unsigned short s[4]; } pk;
  pk.s[0] = f2bf(v.x); pk.s[1] = f2bf(v.y); pk.s[2] = f2bf(v.z); pk.s[3] = f2bf(v.w);
  *(ushort4*)(Wb + (size_t)idx * 4) = pk.s4;
}

// ---------------------------------------------------------------------------
// cvt_tr: fp32 [b][c][t] -> bf16 [b][t][c], 64x64 LDS-tiled transpose,
// coalesced float4 reads and uint4 writes.
// ---------------------------------------------------------------------------
__global__ __launch_bounds__(256)
void cvt_tr(const float* __restrict__ X, unsigned short* __restrict__ Y) {
  __shared__ __align__(16) unsigned short Ld[64 * 72];   // [c][t]
  const int tid = threadIdx.x;
  const int t0 = blockIdx.x * 64, c0 = blockIdx.y * 64, b = blockIdx.z;
#pragma unroll
  for (int ii = 0; ii < 4; ++ii) {
    int idx = ii * 256 + tid;                   // 1024 float4
    int t4 = idx & 15, c = idx >> 4;
    float4 v = *(const float4*)(X + ((size_t)b * C + c0 + c) * T + t0 + t4 * 4);
    union { ushort4 s4; unsigned short s[4]; } pk;
    pk.s[0] = f2bf(v.x); pk.s[1] = f2bf(v.y); pk.s[2] = f2bf(v.z); pk.s[3] = f2bf(v.w);
    *(ushort4*)&Ld[c * 72 + t4 * 4] = pk.s4;
  }
  __syncthreads();
#pragma unroll
  for (int ii = 0; ii < 2; ++ii) {
    int idx = ii * 256 + tid;                   // 512 uint4 (8 c each)
    int c8 = idx & 7, t = idx >> 3;
    union { uint4 q; unsigned short s[8]; } pk;
#pragma unroll
    for (int j = 0; j < 8; ++j) pk.s[j] = Ld[(c8 * 8 + j) * 72 + t];
    *(uint4*)(Y + ((size_t)b * T + t0 + t) * C + c0 + c8 * 8) = pk.q;
  }
}

// ---------------------------------------------------------------------------
// proj: Y = scale*(W @ X) + shift.  64(M) x 128(N) tile, 4 waves,
// 8 WMMAs per 32-K round. BOTH operands bf16, staged via async-to-LDS with
// K-step double buffering (issue i+1 after barrier, compute i).
// Xb: bf16 [b][t][c] ; Wb: bf16 [p][o][c]
// outB: bf16 [b][c][t] ; outBT: bf16 [b][t][c] ; outF: fp32 [b][c][t]
// ---------------------------------------------------------------------------
__global__ __launch_bounds__(128)
void proj_kernel(const unsigned short* __restrict__ Xb,
                 const unsigned short* __restrict__ Wb,
                 const float* __restrict__ scale, const float* __restrict__ shift,
                 int p, unsigned short* __restrict__ outB,
                 unsigned short* __restrict__ outBT, float* __restrict__ outF)
{
  __shared__ __align__(16) unsigned short Wt[2][64 * 40];    // [o][c]
  __shared__ __align__(16) unsigned short Xt[2][128 * 40];   // [t][c]

  const int tid = threadIdx.x;
  const int w = tid >> 5, lane = tid & 31;
  const int n = lane & 15, hf = lane >> 4;
  const int t0 = blockIdx.x * 128;
  const int m0 = blockIdx.y * 64;
  const int b  = blockIdx.z;

  auto stage = [&](int buf, int kk) {
#pragma unroll
    for (int ii = 0; ii < 2; ++ii) {            // W: 64 rows x 64B
      int idx = ii * 128 + tid;
      int c16 = idx & 3, r = idx >> 2;
      async_ld_b128(lds_off32(&Wt[buf][r * 40 + c16 * 8]),
                    Wb + (size_t)p * C * C + (size_t)(m0 + r) * C + kk + c16 * 8);
    }
#pragma unroll
    for (int ii = 0; ii < 4; ++ii) {            // X: 128 rows x 64B
      int idx = ii * 128 + tid;
      int c16 = idx & 3, t = idx >> 2;
      async_ld_b128(lds_off32(&Xt[buf][t * 40 + c16 * 8]),
                    Xb + ((size_t)b * T + t0 + t) * C + kk + c16 * 8);
    }
  };

  v8f acc[8] = {};

  stage(0, 0);
  for (int kk = 0; kk < C; kk += 32) {
    const int buf = (kk >> 5) & 1;
    wait_async0();                // this wave's K-step tiles landed
    __syncthreads();              // everyone's landed; all past prior compute
    if (kk + 32 < C) stage(buf ^ 1, kk + 32);

    v16bf bf0 = load_BT_frag(Xt[buf] + (w * 32) * 40, 40);
    v16bf bf1 = load_BT_frag(Xt[buf] + (w * 32 + 16) * 40, 40);
#pragma unroll
    for (int mt = 0; mt < 4; ++mt) {
      v16bf af = load_A_frag(Wt[buf] + mt * 16 * 40, 40);
      acc[mt * 2 + 0] = WMMA_BF16(af, bf0, acc[mt * 2 + 0]);
      acc[mt * 2 + 1] = WMMA_BF16(af, bf1, acc[mt * 2 + 1]);
    }
  }

#pragma unroll
  for (int mt = 0; mt < 4; ++mt) {
    const float* scp = scale + p * C + m0 + mt * 16 + hf * 8;  // 8 consecutive o
    const float* shp = shift + p * C + m0 + mt * 16 + hf * 8;
    float4 sa = *(const float4*)scp, sb = *(const float4*)(scp + 4);
    float4 ha = *(const float4*)shp, hb = *(const float4*)(shp + 4);
    float sc[8] = {sa.x, sa.y, sa.z, sa.w, sb.x, sb.y, sb.z, sb.w};
    float sh[8] = {ha.x, ha.y, ha.z, ha.w, hb.x, hb.y, hb.z, hb.w};
#pragma unroll
    for (int nt = 0; nt < 2; ++nt) {
      int t = t0 + w * 32 + nt * 16 + n;
      float y[8];
#pragma unroll
      for (int r = 0; r < 8; ++r) y[r] = acc[mt * 2 + nt][r] * sc[r] + sh[r];
      if (outF) {
#pragma unroll
        for (int r = 0; r < 8; ++r) {
          int o = m0 + mt * 16 + r + 8 * hf;
          outF[((size_t)b * C + o) * T + t] = y[r];        // coalesced along t
        }
      }
      if (outB) {
#pragma unroll
        for (int r = 0; r < 8; ++r) {
          int o = m0 + mt * 16 + r + 8 * hf;
          outB[((size_t)b * C + o) * T + t] = f2bf(y[r]);
        }
      }
      if (outBT) {                                         // 8 consecutive o: one 16B store
        union { uint4 q; unsigned short s[8]; } pk;
#pragma unroll
        for (int r = 0; r < 8; ++r) pk.s[r] = f2bf(y[r]);
        *(uint4*)(outBT + ((size_t)b * T + t) * C + m0 + mt * 16 + hf * 8) = pk.q;
      }
    }
  }
}

// ---------------------------------------------------------------------------
// attention: per (b, head, 64 t1-columns), flash over 32-key chunks.
// K/V chunks double-buffered via global_load_async_to_lds_b128.
// ---------------------------------------------------------------------------
__global__ __launch_bounds__(128)
void attn_kernel(const unsigned short* __restrict__ qbt,
                 const unsigned short* __restrict__ kbt,
                 const unsigned short* __restrict__ vb,
                 const unsigned int* __restrict__ mbits,
                 unsigned short* __restrict__ xbt)
{
  __shared__ __align__(16) unsigned short Kt[2][32 * 72];     // [s][d]
  __shared__ __align__(16) unsigned short Vt[2][64 * 40];     // [d][s]
  __shared__ __align__(16) unsigned short Pt[4][16 * 40];     // per wave, [t1][s]

  const int tid = threadIdx.x;
  const int w = tid >> 5, lane = tid & 31;
  const int n = lane & 15, hf = lane >> 4;
  const int b = blockIdx.z, head = blockIdx.y;
  const int t1 = blockIdx.x * 64 + w * 16 + n;
  const size_t cbase = ((size_t)b * C + head * DK) * T;        // [c][t] base (v)
  const size_t tbase = (size_t)b * T * C + head * DK;          // [t][c] base (q,k,x)

  auto stage = [&](int buf, int ch) {
    const int s0 = ch * 32;
#pragma unroll
    for (int ii = 0; ii < 2; ++ii) {                // k^T: 32 rows x 128B
      int idx = ii * 128 + tid;
      int c16 = idx & 7, s = idx >> 3;
      async_ld_b128(lds_off32(&Kt[buf][s * 72 + c16 * 8]),
                    kbt + tbase + (size_t)(s0 + s) * C + c16 * 8);
    }
#pragma unroll
    for (int ii = 0; ii < 2; ++ii) {                // V: 64 rows x 64B
      int idx = ii * 128 + tid;
      int c16 = idx & 3, d = idx >> 2;
      async_ld_b128(lds_off32(&Vt[buf][d * 40 + c16 * 8]),
                    vb + cbase + (size_t)d * T + s0 + c16 * 8);
    }
  };

  v16bf qf[2];
#pragma unroll
  for (int kk = 0; kk < 2; ++kk) {
    Frag16 r;
    const uint4* p = (const uint4*)(qbt + tbase + (size_t)t1 * C + kk * 32 + hf * 16);
    r.q[0] = p[0];
    r.q[1] = p[1];
    qf[kk] = r.v;
  }

  v8f acc[4] = {};
  float m = NEG_BIG, l = 0.0f;
  constexpr int NCH = T / 32;

  stage(0, 0);
  for (int ch = 0; ch < NCH; ++ch) {
    const int buf = ch & 1;
    wait_async0();
    __syncthreads();
    if (ch + 1 < NCH) stage(buf ^ 1, ch + 1);

    v8f S[2];
#pragma unroll
    for (int st = 0; st < 2; ++st) {
      v8f s = {};
#pragma unroll
      for (int kk = 0; kk < 2; ++kk) {
        v16bf a = load_A_frag(Kt[buf] + st * 16 * 72 + kk * 32, 72);
        s = WMMA_BF16(a, qf[kk], s);
      }
      S[st] = s;
    }

    unsigned int mw = mbits[((size_t)b * T + t1) * (T / 32) + ch];

    float cmax = NEG_BIG;
#pragma unroll
    for (int st = 0; st < 2; ++st)
#pragma unroll
      for (int r = 0; r < 8; ++r) {
        int j = st * 16 + r + 8 * hf;
        float sv = ((mw >> j) & 1u) ? S[st][r] * 0.125f : NEG_BIG;
        S[st][r] = sv;
        cmax = fmaxf(cmax, sv);
      }
    cmax = fmaxf(cmax, __shfl_xor(cmax, 16, 32));
    float mnew = fmaxf(m, cmax);
    float alpha = __expf(m - mnew);

    float ls = 0.0f;
#pragma unroll
    for (int st = 0; st < 2; ++st) {
      union { uint4 q; unsigned short s[8]; } pv;
#pragma unroll
      for (int r = 0; r < 8; ++r) {
        int j = st * 16 + r + 8 * hf;
        float p = ((mw >> j) & 1u) ? __expf(S[st][r] - mnew) : 0.0f;
        ls += p;
        pv.s[r] = f2bf(p);
      }
      *(uint4*)&Pt[w][n * 40 + st * 16 + hf * 8] = pv.q;
    }
    ls += __shfl_xor(ls, 16, 32);
    l = l * alpha + ls;
    m = mnew;

#pragma unroll
    for (int mt = 0; mt < 4; ++mt) acc[mt] = acc[mt] * alpha;

    v16bf pf = load_BT_frag(Pt[w], 40);
#pragma unroll
    for (int mt = 0; mt < 4; ++mt) {
      v16bf a = load_A_frag(Vt[buf] + mt * 16 * 40, 40);
      acc[mt] = WMMA_BF16(a, pf, acc[mt]);
    }
  }

  float rl = (l > 0.0f) ? (1.0f / l) : 0.0f;
#pragma unroll
  for (int mt = 0; mt < 4; ++mt) {
    union { uint4 q; unsigned short s[8]; } xv;
#pragma unroll
    for (int r = 0; r < 8; ++r) xv.s[r] = f2bf(acc[mt][r] * rl);
    *(uint4*)(xbt + tbase + (size_t)t1 * C + mt * 16 + hf * 8) = xv.q;
  }
}

// ---------------------------------------------------------------------------
// new_cache = cat(key, value) along channels: (B, 2C, 1, T)
// ---------------------------------------------------------------------------
__global__ void cache_cat(const float* __restrict__ key, const float* __restrict__ val,
                          float* __restrict__ out) {
  size_t i4 = ((size_t)blockIdx.x * 256 + threadIdx.x) * 4;
  const size_t per = (size_t)C * T;
  if (i4 >= (size_t)B * 2 * per) return;
  size_t b = i4 / (2 * per), r = i4 % (2 * per);
  const float* src = (r < per) ? (key + b * per + r) : (val + b * per + (r - per));
  float4 v = *(const float4*)src;
  *(float4*)(out + i4) = v;
}

// ---------------------------------------------------------------------------
extern "C" void kernel_launch(void* const* d_in, const int* in_sizes, int n_in,
                              void* d_out, int out_size, void* d_ws, size_t ws_size,
                              hipStream_t stream) {
  (void)in_sizes; (void)n_in; (void)out_size; (void)ws_size;
  const float* query = (const float*)d_in[0];
  const float* key   = (const float*)d_in[1];
  const float* value = (const float*)d_in[2];
  const int*   mask  = (const int*)d_in[3];
  const float* Ws    = (const float*)d_in[4];
  const float* bs    = (const float*)d_in[5];
  const float* gam   = (const float*)d_in[6];
  const float* bet   = (const float*)d_in[7];
  const float* mean  = (const float*)d_in[8];
  const float* var   = (const float*)d_in[9];
  float* out = (float*)d_out;

  const size_t bct = (size_t)B * C * T;
  unsigned short* q_bt = (unsigned short*)d_ws;            // [b][t][c]
  unsigned short* k_bt = q_bt + bct;                       // [b][t][c]
  unsigned short* v_bf = k_bt + bct;                       // [b][c][t]
  unsigned short* x_bt = v_bf + bct;                       // [b][t][c]
  unsigned short* in_bf = x_bt + bct;                      // [b][t][c], reused q/k/v input
  unsigned short* w_bf  = in_bf + bct;                     // [p][o][c]
  unsigned int*   mbit  = (unsigned int*)(w_bf + (size_t)4 * C * C);
  float*          bnsc  = (float*)(mbit + (size_t)B * T * (T / 32));
  float*          bnsh  = bnsc + 4 * C;

  bn_fold<<<(4 * C + 255) / 256, 256, 0, stream>>>(bs, gam, bet, mean, var, bnsc, bnsh);
  mask_pack<<<(B * T * (T / 32) + 255) / 256, 256, 0, stream>>>(mask, mbit);
  cvt_w<<<(4 * C * C / 4 + 255) / 256, 256, 0, stream>>>(Ws, w_bf);

  dim3 cg(T / 64, C / 64, B);
  dim3 pg(T / 128, C / 64, B);
  cvt_tr<<<cg, 256, 0, stream>>>(query, in_bf);
  proj_kernel<<<pg, 128, 0, stream>>>(in_bf, w_bf, bnsc, bnsh, 0, nullptr, q_bt, nullptr);
  cvt_tr<<<cg, 256, 0, stream>>>(key, in_bf);
  proj_kernel<<<pg, 128, 0, stream>>>(in_bf, w_bf, bnsc, bnsh, 1, nullptr, k_bt, nullptr);
  cvt_tr<<<cg, 256, 0, stream>>>(value, in_bf);
  proj_kernel<<<pg, 128, 0, stream>>>(in_bf, w_bf, bnsc, bnsh, 2, v_bf, nullptr, nullptr);

  attn_kernel<<<dim3(T / 64, H, B), 128, 0, stream>>>(q_bt, k_bt, v_bf, mbit, x_bt);
  proj_kernel<<<pg, 128, 0, stream>>>(x_bt, w_bf, bnsc, bnsh, 3, nullptr, nullptr, out);

  {
    size_t n4 = (size_t)B * 2 * C * T / 4;
    cache_cat<<<(unsigned)((n4 + 255) / 256), 256, 0, stream>>>(key, value, out + bct);
  }
}
